// TAB_79654463471808
// MI455X (gfx1250) — compile-verified, matrix operand order checked by reading, not run on hardware
//
// MI455X (gfx1250) implementation of the TAB block.
// All dense GEMMs + both attention matmuls use v_wmma_f32_16x16x32_bf16
// (fp32 accumulate). Working set ~200MB -> L2 resident; compute-bound on WMMA.
// GEMM A-panel staging uses GLOBAL_LOAD_ASYNC_TO_LDS_B128 (ASYNCcnt) via
// inline asm, per CDNA5 ISA.
#include <hip/hip_runtime.h>
#include <math.h>

#define USE_ASYNC_ASM 1

#define BATCH 2
#define DIM   192
#define QD    192
#define MP    8
#define MDIM  384
#define HH    128
#define WW    128
#define NN    (HH*WW)        /* 16384 tokens per batch */
#define NT    (BATCH*NN)     /* 32768 */
#define NHEAD 6
#define DHEAD 32
#define GS    128
#define PSTR  296            /* padded LDS row stride (elements) in attention */

typedef __bf16 v16bf __attribute__((ext_vector_type(16)));
typedef float  v8f   __attribute__((ext_vector_type(8)));

struct FragBF { union { v16bf v; uint4 q[2]; unsigned short u[16]; }; };
struct FragF  { union { v8f  v; float f[8]; }; };

__device__ __forceinline__ unsigned short f2bf(float x) {
    unsigned u = __float_as_uint(x);
    unsigned r = u + 0x7FFFu + ((u >> 16) & 1u);   // round-to-nearest-even
    return (unsigned short)(r >> 16);
}
__device__ __forceinline__ float gelu(float x) {
    return 0.5f * x * (1.0f + erff(x * 0.70710678118f));
}
__device__ __forceinline__ float wredsum(float v){
    #pragma unroll
    for (int o = 16; o; o >>= 1) v += __shfl_xor(v, o, 32);
    return v;
}
__device__ __forceinline__ float hredsum(float v){   // within 16-lane halves
    #pragma unroll
    for (int o = 8; o; o >>= 1) v += __shfl_xor(v, o, 32);
    return v;
}
__device__ __forceinline__ float hredmax(float v){
    #pragma unroll
    for (int o = 8; o; o >>= 1) v = fmaxf(v, __shfl_xor(v, o, 32));
    return v;
}

// Load a 16x32 bf16 fragment from an LDS panel laid out row-major
// (rows = matrix rows for A, = output columns for B-transposed).
// Matches ISA layout: lane L: row = L&15, k-half = L>>4;
// VGPR0-3 hold k = 8h..8h+7, VGPR4-7 hold k = 16+8h..23+8h  -> two b128 loads.
__device__ __forceinline__ FragBF lds_frag(const unsigned short* base, int stride,
                                           int ks, int lane){
    int m = lane & 15, h = lane >> 4;
    const unsigned short* p = base + m * stride + ks + 8 * h;
    FragBF f;
    f.q[0] = *(const uint4*)p;
    f.q[1] = *(const uint4*)(p + 16);
    return f;
}

#if USE_ASYNC_ASM
// Direct global -> LDS async copy (16B per lane), tracked by ASYNCcnt.
// LDS byte address = low 32 bits of the flat pointer (aperture truncation,
// CDNA5 ISA aperture rules). Operands per §15.18.3 op 98.
__device__ __forceinline__ void async_g2l_b128(const void* gsrc, void* ldst){
    unsigned       laddr = (unsigned)(size_t)ldst;
    unsigned long long ga = (unsigned long long)(size_t)gsrc;
    asm volatile("global_load_async_to_lds_b128 %0, %1, off"
                 :: "v"(laddr), "v"(ga) : "memory");
}
__device__ __forceinline__ void async_wait0(){
    asm volatile("s_wait_asynccnt 0x0" ::: "memory");
}
#endif

// ---------------------------------------------------------------- GEMM (WMMA)
// C[M,N] = act(A[M,K] @ B[K,N] + bias) (+resid). A,B bf16; C fp32 and/or bf16.
// Block tile 32x64, 8 waves (2x4 grid of 16x16 WMMA tiles), K chunked by 64.
#define GKC  64
#define LDAS 72
__global__ __launch_bounds__(256)
void k_gemm(const unsigned short* __restrict__ A, const unsigned short* __restrict__ Bw,
            const float* __restrict__ bias, const float* __restrict__ resid,
            float* __restrict__ Cf, unsigned short* __restrict__ Cb,
            int M, int N, int K, int act)
{
    (void)M;
    __shared__ unsigned short As[32][LDAS];
    __shared__ unsigned short Bs[64][LDAS];   // stored transposed: Bs[n][k]
    const int tid = threadIdx.x;
    const int lane = tid & 31, wave = tid >> 5;
    const int wm = wave >> 2, wn = wave & 3;
    const int row0 = blockIdx.y * 32, col0 = blockIdx.x * 64;
    FragF acc;
    #pragma unroll
    for (int r = 0; r < 8; r++) acc.f[r] = 0.f;

    for (int k0 = 0; k0 < K; k0 += GKC) {
        {   // stage A chunk (32x64): async global->LDS, 8 bf16 (16B) per lane
            int r = tid >> 3, c = (tid & 7) * 8;
            const unsigned short* g = A + (size_t)(row0 + r) * K + k0 + c;
#if USE_ASYNC_ASM
            async_g2l_b128(g, &As[r][c]);
#else
            *(uint4*)&As[r][c] = *(const uint4*)g;
#endif
            if (k0 + GKC < K)
                __builtin_prefetch(A + (size_t)(row0 + r) * K + k0 + GKC + c, 0, 1);
        }
        {   // stage B chunk (64x64) with transpose into Bs[n][k]
            int kk = tid >> 2, ng = (tid & 3) * 16;
            const unsigned short* src = Bw + (size_t)(k0 + kk) * N + col0 + ng;
            __align__(16) unsigned short tb[16];
            *(uint4*)tb       = *(const uint4*)src;
            *(uint4*)(tb + 8) = *(const uint4*)(src + 8);
            #pragma unroll
            for (int j = 0; j < 16; j++) Bs[ng + j][kk] = tb[j];
        }
#if USE_ASYNC_ASM
        async_wait0();                // A panel landed in LDS
#endif
        __syncthreads();
        #pragma unroll
        for (int ks = 0; ks < GKC; ks += 32) {
            FragBF fa = lds_frag(&As[wm * 16][0], LDAS, ks, lane);
            FragBF fb = lds_frag(&Bs[wn * 16][0], LDAS, ks, lane);
            acc.v = __builtin_amdgcn_wmma_f32_16x16x32_bf16(false, fa.v, false, fb.v,
                                                            (short)0, acc.v, false, false);
        }
        __syncthreads();
    }
    const int ncol = lane & 15, hh = lane >> 4;
    const int col = col0 + wn * 16 + ncol;
    #pragma unroll
    for (int r = 0; r < 8; r++) {
        int row = row0 + wm * 16 + r + 8 * hh;
        float v = acc.f[r];
        if (bias)  v += bias[col];
        if (act)   v = gelu(v);
        if (resid) v += resid[(size_t)row * N + col];
        size_t o = (size_t)row * N + col;
        if (Cf) Cf[o] = v;
        if (Cb) Cb[o] = f2bf(v);
    }
}

// --------------------------------------------------------- tiled transpose
// src [R][C] -> dst [C][R], per batch (blockIdx.z).
__global__ __launch_bounds__(256)
void k_transpose(const float* __restrict__ src, float* __restrict__ dst, int R, int C)
{
    __shared__ float tile[32][33];
    int b = blockIdx.z;
    const float* s = src + (size_t)b * R * C;
    float* d = dst + (size_t)b * R * C;
    int c0 = blockIdx.x * 32, r0 = blockIdx.y * 32;
    #pragma unroll
    for (int i = 0; i < 4; i++) {
        int r = r0 + threadIdx.y + i * 8;
        tile[threadIdx.y + i * 8][threadIdx.x] = s[(size_t)r * C + c0 + threadIdx.x];
    }
    __syncthreads();
    #pragma unroll
    for (int i = 0; i < 4; i++) {
        int c = c0 + threadIdx.y + i * 8;
        d[(size_t)c * R + r0 + threadIdx.x] = tile[threadIdx.x][threadIdx.y + i * 8];
    }
}

// --------------------------------------------------------------- LayerNorm
// wave per row (D=192 -> 6 elems/lane), fp32 and/or bf16 out.
__global__ __launch_bounds__(256)
void k_ln(const float* __restrict__ src, const float* __restrict__ g,
          const float* __restrict__ bb, float* __restrict__ outf,
          unsigned short* __restrict__ outb, int rows)
{
    int lane = threadIdx.x & 31;
    int row = blockIdx.x * 8 + (threadIdx.x >> 5);
    if (row >= rows) return;
    const float* p = src + (size_t)row * DIM;
    float v[6]; float s = 0.f, s2 = 0.f;
    #pragma unroll
    for (int j = 0; j < 6; j++) { int c = lane + 32 * j; v[j] = p[c]; s += v[j]; s2 += v[j] * v[j]; }
    s = wredsum(s); s2 = wredsum(s2);
    float mean = s * (1.f / DIM);
    float var  = s2 * (1.f / DIM) - mean * mean;
    float inv  = rsqrtf(var + 1e-5f);
    #pragma unroll
    for (int j = 0; j < 6; j++) {
        int c = lane + 32 * j;
        float o = (v[j] - mean) * inv * g[c] + bb[c];
        if (outf) outf[(size_t)row * DIM + c] = o;
        if (outb) outb[(size_t)row * DIM + c] = f2bf(o);
    }
}

// ------------------------------------------------------------- converters
__global__ void k_f2b(const float* __restrict__ s, unsigned short* __restrict__ d, int n){
    for (int i = blockIdx.x * 256 + threadIdx.x; i < n; i += gridDim.x * 256) d[i] = f2bf(s[i]);
}
__global__ void k_wtrans(const float* __restrict__ s, unsigned short* __restrict__ d){
    int id = blockIdx.x * 256 + threadIdx.x;
    if (id >= DIM * DIM) return;
    int o = id / DIM, kk = id % DIM;
    d[(size_t)kk * DIM + o] = f2bf(s[(size_t)o * DIM + kk]);   // conv1_w^T
}
__global__ void k_zero(float* p, int n){
    int i = blockIdx.x * 256 + threadIdx.x; if (i < n) p[i] = 0.f;
}

// ----------------------------------------------- DPR: assignment + proto sums
// pacc layout: [b][m][0..191]=sum(a*xn), [b][m][192]=sum(a)
__global__ __launch_bounds__(256)
void k_assign(const float* __restrict__ e, const float* __restrict__ xn,
              const float* __restrict__ aw, float* __restrict__ pacc)
{
    __shared__ float as8[256][MP];
    int b = blockIdx.y, tid = threadIdx.x;
    int tok0 = blockIdx.x * 256;
    const float* er = e + ((size_t)b * NN + tok0 + tid) * DIM;
    float l[MP];
    #pragma unroll
    for (int m = 0; m < MP; m++) l[m] = 0.f;
    for (int c = 0; c < DIM; c++) {
        float ev = er[c];
        #pragma unroll
        for (int m = 0; m < MP; m++) l[m] += ev * aw[c * MP + m];
    }
    float mx = l[0];
    #pragma unroll
    for (int m = 1; m < MP; m++) mx = fmaxf(mx, l[m]);
    float sm = 0.f;
    #pragma unroll
    for (int m = 0; m < MP; m++) { l[m] = expf(l[m] - mx); sm += l[m]; }
    float inv = 1.f / sm;
    #pragma unroll
    for (int m = 0; m < MP; m++) as8[tid][m] = l[m] * inv;
    __syncthreads();
    for (int slot = tid; slot < MP * 193; slot += 256) {
        int m = slot / 193, c = slot % 193;
        float acc = 0.f;
        if (c == 192) {
            for (int t = 0; t < 256; t++) acc += as8[t][m];
        } else {
            const float* xr = xn + ((size_t)b * NN + tok0) * DIM + c;
            for (int t = 0; t < 256; t++) acc += as8[t][m] * xr[(size_t)t * DIM];
        }
        atomicAdd(&pacc[b * (MP * 193) + slot], acc);
    }
}

// proto0 = l2n(ln(num/den)); qp=(proto0+pq)@rq_w; s_rk = qp @ rk_w^T
__global__ __launch_bounds__(256)
void k_proto1(const float* __restrict__ pacc, const float* __restrict__ pg,
              const float* __restrict__ pb, const float* __restrict__ pq,
              const float* __restrict__ rq_w, const float* __restrict__ rk_w,
              float* __restrict__ proto0, float* __restrict__ s_rk)
{
    __shared__ float sp[MP][DIM];
    __shared__ float sq[MP][DIM];
    int b = blockIdx.x, lane = threadIdx.x & 31, m = threadIdx.x >> 5;
    float den = fmaxf(pacc[b * (MP * 193) + m * 193 + 192], 1e-6f);
    float v[6]; float s = 0.f;
    #pragma unroll
    for (int j = 0; j < 6; j++) {
        int c = lane + 32 * j;
        v[j] = pacc[b * (MP * 193) + m * 193 + c] / den; s += v[j];
    }
    s = wredsum(s);
    float mean = s * (1.f / DIM);
    float s2 = 0.f;
    #pragma unroll
    for (int j = 0; j < 6; j++) { float t = v[j] - mean; s2 += t * t; }
    s2 = wredsum(s2);
    float inv = rsqrtf(s2 * (1.f / DIM) + 1e-5f);
    float ss = 0.f;
    #pragma unroll
    for (int j = 0; j < 6; j++) {
        int c = lane + 32 * j;
        v[j] = (v[j] - mean) * inv * pg[c] + pb[c]; ss += v[j] * v[j];
    }
    ss = wredsum(ss);
    float nr = 1.f / fmaxf(sqrtf(ss), 1e-12f);
    #pragma unroll
    for (int j = 0; j < 6; j++) {
        int c = lane + 32 * j; v[j] *= nr;
        sp[m][c] = v[j];
        proto0[(size_t)(b * MP + m) * DIM + c] = v[j];
    }
    __syncthreads();
    #pragma unroll
    for (int j = 0; j < 6; j++) {           // qp
        int qd = lane + 32 * j; float acc = 0.f;
        for (int c = 0; c < DIM; c++) acc += (sp[m][c] + pq[m * DIM + c]) * rq_w[c * QD + qd];
        sq[m][qd] = acc;
    }
    __syncthreads();
    #pragma unroll
    for (int j = 0; j < 6; j++) {           // s_rk[m][c] = qp . rk_w[c,:]
        int c = lane + 32 * j; float acc = 0.f;
        for (int d = 0; d < QD; d++) acc += sq[m][d] * rk_w[c * QD + d];
        s_rk[(size_t)(b * MP + m) * DIM + c] = acc;
    }
}

// rattn softmax over n=16384 per (b,m); rsum = rattn @ xn
__global__ __launch_bounds__(192)
void k_rattn(const float* __restrict__ s_rk, const float* __restrict__ e,
             const float* __restrict__ xn, float* __restrict__ rlog,
             float* __restrict__ rsum, float scale)
{
    __shared__ float red[192];
    __shared__ float stats[2];
    int bm = blockIdx.x; int b = bm >> 3, m = bm & 7;
    int tid = threadIdx.x;
    const float* sr = s_rk + (size_t)(b * MP + m) * DIM;
    float* lg = rlog + (size_t)(b * MP + m) * NN;
    float lmax = -3e38f;
    for (int i = tid; i < NN; i += 192) {
        const float* er = e + ((size_t)b * NN + i) * DIM;
        float acc = 0.f;
        for (int c = 0; c < DIM; c++) acc += sr[c] * er[c];
        float l = acc * scale; lg[i] = l; lmax = fmaxf(lmax, l);
    }
    red[tid] = lmax; __syncthreads();
    if (tid < 64) red[tid] = fmaxf(red[tid], fmaxf(red[tid + 64], red[tid + 128]));
    __syncthreads();
    for (int sft = 32; sft > 0; sft >>= 1) {
        if (tid < sft) red[tid] = fmaxf(red[tid], red[tid + sft]);
        __syncthreads();
    }
    if (tid == 0) stats[0] = red[0];
    __syncthreads();
    float mx = stats[0];
    float lsum = 0.f;
    for (int i = tid; i < NN; i += 192) lsum += expf(lg[i] - mx);
    red[tid] = lsum; __syncthreads();
    if (tid < 64) red[tid] = red[tid] + red[tid + 64] + red[tid + 128];
    __syncthreads();
    for (int sft = 32; sft > 0; sft >>= 1) {
        if (tid < sft) red[tid] += red[tid + sft];
        __syncthreads();
    }
    if (tid == 0) stats[1] = red[0];
    __syncthreads();
    float inv = 1.f / stats[1];
    int c = tid;                              // thread owns one channel
    float acc = 0.f;
    for (int i = 0; i < NN; i++) {
        float w = expf(lg[i] - mx) * inv;
        acc += w * xn[((size_t)b * NN + i) * DIM + c];
    }
    rsum[(size_t)(b * MP + m) * DIM + c] = acc;
}

// proto2 = l2n(ln(proto0 + sig(gate)*(rsum@rv_w))); pf, Pk, Pv projections
__global__ __launch_bounds__(256)
void k_proto2(const float* __restrict__ proto0, const float* __restrict__ rsum,
              const float* __restrict__ rv_w, const float* __restrict__ pg,
              const float* __restrict__ pb, const float* __restrict__ gate,
              const float* __restrict__ pproj_w, const float* __restrict__ pk_w,
              const float* __restrict__ pv_w, float* __restrict__ pf,
              unsigned short* __restrict__ Pkb, unsigned short* __restrict__ Pvb)
{
    __shared__ float sp[MP][DIM];
    int b = blockIdx.x, lane = threadIdx.x & 31, m = threadIdx.x >> 5;
    float sg = 1.f / (1.f + expf(-gate[0]));
    const float* rs = rsum + (size_t)(b * MP + m) * DIM;
    float v[6]; float s = 0.f;
    #pragma unroll
    for (int j = 0; j < 6; j++) {
        int c = lane + 32 * j;
        float acc = 0.f;
        for (int c1 = 0; c1 < DIM; c1++) acc += rs[c1] * rv_w[c1 * DIM + c];
        v[j] = proto0[(size_t)(b * MP + m) * DIM + c] + sg * acc; s += v[j];
    }
    s = wredsum(s);
    float mean = s * (1.f / DIM);
    float s2 = 0.f;
    #pragma unroll
    for (int j = 0; j < 6; j++) { float t = v[j] - mean; s2 += t * t; }
    s2 = wredsum(s2);
    float inv = rsqrtf(s2 * (1.f / DIM) + 1e-5f);
    float ss = 0.f;
    #pragma unroll
    for (int j = 0; j < 6; j++) {
        int c = lane + 32 * j;
        v[j] = (v[j] - mean) * inv * pg[c] + pb[c]; ss += v[j] * v[j];
    }
    ss = wredsum(ss);
    float nr = 1.f / fmaxf(sqrtf(ss), 1e-12f);
    #pragma unroll
    for (int j = 0; j < 6; j++) { int c = lane + 32 * j; v[j] *= nr; sp[m][c] = v[j]; }
    __syncthreads();
    float w[6]; float ps = 0.f;
    #pragma unroll
    for (int j = 0; j < 6; j++) {            // pf = l2n(proto2 @ pproj_w)
        int qd = lane + 32 * j; float acc = 0.f;
        for (int c = 0; c < DIM; c++) acc += sp[m][c] * pproj_w[c * QD + qd];
        w[j] = acc; ps += acc * acc;
    }
    ps = wredsum(ps);
    float pn = 1.f / fmaxf(sqrtf(ps), 1e-12f);
    #pragma unroll
    for (int j = 0; j < 6; j++) {
        int qd = lane + 32 * j;
        pf[(size_t)(b * MP + m) * QD + qd] = w[j] * pn;
    }
    #pragma unroll
    for (int j = 0; j < 6; j++) {            // Pk / Pv (bf16 for attention)
        int jd = lane + 32 * j;
        float ak = 0.f, av = 0.f;
        for (int c = 0; c < DIM; c++) {
            ak += sp[m][c] * pk_w[c * QD + jd];
            av += sp[m][c] * pv_w[c * DIM + jd];
        }
        Pkb[(size_t)(b * MP + m) * DIM + jd] = f2bf(ak);
        Pvb[(size_t)(b * MP + m) * DIM + jd] = f2bf(av);
    }
}

// routing scores -> sort key packed u64 = (float_bits(key)<<32)|token
__global__ __launch_bounds__(256)
void k_route(const float* __restrict__ tf, const float* __restrict__ pf,
             unsigned long long* __restrict__ kp, float scale)
{
    int b = blockIdx.y;
    int i = blockIdx.x * 256 + threadIdx.x;
    const float* tr = tf + ((size_t)b * NN + i) * QD;
    float ss = 0.f;
    for (int d = 0; d < QD; d++) { float t = tr[d]; ss += t * t; }
    float nr = 1.f / fmaxf(sqrtf(ss), 1e-12f);
    float l[MP];
    #pragma unroll
    for (int m = 0; m < MP; m++) l[m] = 0.f;
    for (int d = 0; d < QD; d++) {
        float t = tr[d] * nr;
        #pragma unroll
        for (int m = 0; m < MP; m++) l[m] += t * pf[(size_t)(b * MP + m) * QD + d];
    }
    int am = 0; float mx = l[0] * scale;
    #pragma unroll
    for (int m = 0; m < MP; m++) { l[m] *= scale; if (l[m] > mx) { mx = l[m]; am = m; } }
    float sm = 0.f;
    #pragma unroll
    for (int m = 0; m < MP; m++) sm += expf(l[m] - mx);
    float maxp = 1.f / sm;                         // softmax value at argmax
    float key = (float)am + 0.5f * (1.f - maxp);   // key >= 0 -> bits monotone
    kp[(size_t)b * NN + i] = ((unsigned long long)__float_as_uint(key) << 32) | (unsigned)i;
}

__global__ void k_bitonic(unsigned long long* kp, int j, int kk){
    int id = blockIdx.x * 256 + threadIdx.x;
    int b = id >> 14;            // / NN
    int i = id & (NN - 1);
    int ixj = i ^ j;
    if (ixj <= i) return;
    unsigned long long* base = kp + ((size_t)b << 14);
    unsigned long long a = base[i], c = base[ixj];
    bool up = ((i & kk) == 0);
    bool sw = up ? (a > c) : (a < c);
    if (sw) { base[i] = c; base[ixj] = a; }
}

__global__ void k_gather(const unsigned long long* __restrict__ kp,
                         const unsigned short* __restrict__ xnb,
                         unsigned short* __restrict__ sxb){
    int id = blockIdx.x * 256 + threadIdx.x;      // NT*24 (8-elem chunks)
    int row = id / 24, ch = id % 24;
    int b = row >> 14, i = row & (NN - 1);
    unsigned src = (unsigned)(kp[(size_t)b * NN + i] & 0xFFFFFFFFull);
    *(uint4*)(sxb + (size_t)row * DIM + ch * 8) =
        *(const uint4*)(xnb + ((size_t)b * NN + src) * DIM + ch * 8);
}

__global__ void k_scatter(const unsigned long long* __restrict__ kp,
                          const float* __restrict__ outS,
                          unsigned short* __restrict__ outb){
    int id = blockIdx.x * 256 + threadIdx.x;      // NT*DIM
    int c = id % DIM; int row = id / DIM;
    int b = row >> 14, i = row & (NN - 1);
    unsigned dst = (unsigned)(kp[(size_t)b * NN + i] & 0xFFFFFFFFull);
    outb[((size_t)b * NN + dst) * DIM + c] = f2bf(outS[(size_t)row * DIM + c]);
}

// ------------------------------------------------- IASA attention (WMMA)
// One workgroup per (window g, head, batch); 4 waves, wave = one 16-row q tile
// per iteration (2 iters). 17 logit WMMAs (256-key window + proto tile),
// register softmax with half-wave shuffles, p -> LDS -> 18 output WMMAs.
__global__ __launch_bounds__(128)
void k_attn(const unsigned short* __restrict__ qb, const unsigned short* __restrict__ kb,
            const unsigned short* __restrict__ vb, const unsigned short* __restrict__ Pkb,
            const unsigned short* __restrict__ Pvb, float* __restrict__ outS, float sc)
{
    __shared__ unsigned short vT[32][PSTR];       // V^T panel (d-major)
    __shared__ unsigned short pbuf[4][16][PSTR];  // per-wave probability tiles
    const int g = blockIdx.x, head = blockIdx.y, b = blockIdx.z;
    const int tid = threadIdx.x, lane = tid & 31, wave = tid >> 5;

    for (int kj = tid; kj < PSTR; kj += 128) {    // stage transposed V window
        __align__(16) unsigned short tmp[32];
        if (kj < 256) {
            int basei = g * GS + kj;
            int krow = (basei < NN) ? basei : (2 * NN - 1 - basei);   // mirror pad
            const uint4* p = (const uint4*)(vb + ((size_t)b * NN + krow) * DIM + head * DHEAD);
            *(uint4*)(tmp) = p[0]; *(uint4*)(tmp + 8)  = p[1];
            *(uint4*)(tmp + 16) = p[2]; *(uint4*)(tmp + 24) = p[3];
        } else if (kj < 256 + MP) {
            const uint4* p = (const uint4*)(Pvb + ((size_t)(b * MP + (kj - 256))) * DIM + head * DHEAD);
            *(uint4*)(tmp) = p[0]; *(uint4*)(tmp + 8)  = p[1];
            *(uint4*)(tmp + 16) = p[2]; *(uint4*)(tmp + 24) = p[3];
        } else {
            #pragma unroll
            for (int d = 0; d < 32; d++) tmp[d] = 0;
        }
        #pragma unroll
        for (int d = 0; d < 32; d++) vT[d][kj] = tmp[d];
    }
    for (int idx = tid; idx < 4 * 16 * (PSTR - 272); idx += 128) {   // zero p pad
        int w = idx / (16 * (PSTR - 272));
        int rem = idx % (16 * (PSTR - 272));
        pbuf[w][rem / (PSTR - 272)][272 + rem % (PSTR - 272)] = 0;
    }
    __syncthreads();

    const int ncol = lane & 15, hh = lane >> 4;
    for (int it = 0; it < 2; it++) {
        const int mTile = it * 4 + wave;
        FragBF fq;
        {
            const unsigned short* p = qb + ((size_t)b * NN + g * GS + mTile * 16 + ncol) * DIM
                                         + head * DHEAD + 8 * hh;
            fq.q[0] = *(const uint4*)p; fq.q[1] = *(const uint4*)(p + 16);
        }
        FragF t[17];
        #pragma unroll
        for (int nt = 0; nt < 17; nt++) {
            #pragma unroll
            for (int r = 0; r < 8; r++) t[nt].f[r] = 0.f;
            FragBF fk;
            if (nt < 16) {
                int basei = g * GS + nt * 16 + ncol;
                int krow = (basei < NN) ? basei : (2 * NN - 1 - basei);
                const unsigned short* p = kb + ((size_t)b * NN + krow) * DIM + head * DHEAD + 8 * hh;
                fk.q[0] = *(const uint4*)p; fk.q[1] = *(const uint4*)(p + 16);
            } else if (ncol < MP) {
                const unsigned short* p = Pkb + ((size_t)(b * MP + ncol)) * DIM + head * DHEAD + 8 * hh;
                fk.q[0] = *(const uint4*)p; fk.q[1] = *(const uint4*)(p + 16);
            } else {
                #pragma unroll
                for (int z = 0; z < 16; z++) fk.u[z] = 0;
            }
            t[nt].v = __builtin_amdgcn_wmma_f32_16x16x32_bf16(false, fq.v, false, fk.v,
                                                              (short)0, t[nt].v, false, false);
        }
        // two independent softmaxes (window a1, proto a2); outputs sum.
        #pragma unroll
        for (int r = 0; r < 8; r++) {
            float m1 = -3e38f;
            #pragma unroll
            for (int nt = 0; nt < 16; nt++) { t[nt].f[r] *= sc; m1 = fmaxf(m1, t[nt].f[r]); }
            m1 = hredmax(m1);
            float s1 = 0.f;
            #pragma unroll
            for (int nt = 0; nt < 16; nt++) { float p = expf(t[nt].f[r] - m1); t[nt].f[r] = p; s1 += p; }
            s1 = hredsum(s1);
            float inv1 = 1.f / s1;
            #pragma unroll
            for (int nt = 0; nt < 16; nt++) t[nt].f[r] *= inv1;
            float lp = (ncol < MP) ? t[16].f[r] * sc : -3e38f;
            float m2 = hredmax(lp);
            float p2 = expf(lp - m2);
            float s2 = hredsum(p2);
            t[16].f[r] = p2 / s2;
        }
        #pragma unroll
        for (int nt = 0; nt < 17; nt++) {
            #pragma unroll
            for (int r = 0; r < 8; r++)
                pbuf[wave][r + 8 * hh][nt * 16 + ncol] = f2bf(t[nt].f[r]);
        }
        __syncthreads();
        FragF o0, o1;
        #pragma unroll
        for (int r = 0; r < 8; r++) { o0.f[r] = 0.f; o1.f[r] = 0.f; }
        #pragma unroll
        for (int ks = 0; ks < 288; ks += 32) {
            FragBF fp  = lds_frag(&pbuf[wave][0][0], PSTR, ks, lane);
            FragBF fv0 = lds_frag(&vT[0][0],  PSTR, ks, lane);
            FragBF fv1 = lds_frag(&vT[16][0], PSTR, ks, lane);
            o0.v = __builtin_amdgcn_wmma_f32_16x16x32_bf16(false, fp.v, false, fv0.v,
                                                           (short)0, o0.v, false, false);
            o1.v = __builtin_amdgcn_wmma_f32_16x16x32_bf16(false, fp.v, false, fv1.v,
                                                           (short)0, o1.v, false, false);
        }
        #pragma unroll
        for (int r = 0; r < 8; r++) {
            size_t row = (size_t)b * NN + g * GS + mTile * 16 + r + 8 * hh;
            outS[row * DIM + head * DHEAD + ncol]      = o0.f[r];
            outS[row * DIM + head * DHEAD + 16 + ncol] = o1.f[r];
        }
        __syncthreads();
    }
}

// ------------------------------------------------- 5x5 depthwise + gelu + add
__global__ __launch_bounds__(256)
void k_dwconv(const float* __restrict__ z, const float* __restrict__ dww,
              const float* __restrict__ dwb, unsigned short* __restrict__ sb)
{
    int id = blockIdx.x * 256 + threadIdx.x;      // BATCH*NN*MDIM
    int ch = id % MDIM;
    int pix = (id / MDIM) % NN;
    int b = id / (MDIM * NN);
    int y = pix >> 7, x = pix & 127;
    float acc = 0.f;
    #pragma unroll
    for (int dy = -2; dy <= 2; dy++) {
        int yy = y + dy; if (yy < 0 || yy >= HH) continue;
        #pragma unroll
        for (int dx = -2; dx <= 2; dx++) {
            int xx = x + dx; if (xx < 0 || xx >= WW) continue;
            acc += z[((size_t)b * NN + yy * WW + xx) * MDIM + ch]
                 * dww[ch * 25 + (dy + 2) * 5 + (dx + 2)];
        }
    }
    float zc = gelu(acc + dwb[ch]);
    float val = z[((size_t)b * NN + pix) * MDIM + ch] + zc;
    sb[(size_t)id] = f2bf(val);
}

// =========================================================== host dispatch
extern "C" void kernel_launch(void* const* d_in, const int* in_sizes, int n_in,
                              void* d_out, int out_size, void* d_ws, size_t ws_size,
                              hipStream_t stream)
{
    (void)in_sizes; (void)n_in; (void)out_size; (void)ws_size;
    const float* x        = (const float*)d_in[0];
    const float* norm_g   = (const float*)d_in[1];
    const float* norm_b   = (const float*)d_in[2];
    const float* embln_g  = (const float*)d_in[3];
    const float* embln_b  = (const float*)d_in[4];
    const float* emb_w    = (const float*)d_in[5];
    const float* emb_b    = (const float*)d_in[6];
    const float* assign_w = (const float*)d_in[7];
    const float* pq       = (const float*)d_in[8];
    const float* rq_w     = (const float*)d_in[9];
    const float* rk_w     = (const float*)d_in[10];
    const float* rv_w     = (const float*)d_in[11];
    const float* tproj_w  = (const float*)d_in[12];
    const float* pproj_w  = (const float*)d_in[13];
    const float* pln_g    = (const float*)d_in[14];
    const float* pln_b    = (const float*)d_in[15];
    const float* gate     = (const float*)d_in[16];
    const float* q_w      = (const float*)d_in[17];
    const float* k_w      = (const float*)d_in[18];
    const float* v_w      = (const float*)d_in[19];
    const float* pk_w     = (const float*)d_in[20];
    const float* pv_w     = (const float*)d_in[21];
    const float* proj_w   = (const float*)d_in[22];
    const float* conv1_w  = (const float*)d_in[23];
    const float* mlpln_g  = (const float*)d_in[24];
    const float* mlpln_b  = (const float*)d_in[25];
    const float* fc1_w    = (const float*)d_in[26];
    const float* fc1_b    = (const float*)d_in[27];
    const float* dw_w     = (const float*)d_in[28];
    const float* dw_b     = (const float*)d_in[29];
    const float* fc2_w    = (const float*)d_in[30];
    const float* fc2_b    = (const float*)d_in[31];
    float* out            = (float*)d_out;

    char* ws = (char*)d_ws;
    size_t off = 0;
    auto alloc = [&](size_t bytes) -> char* {
        char* p = ws + off;
        off += (bytes + 255) & ~(size_t)255;
        return p;
    };
    const size_t SF = (size_t)NT * DIM * sizeof(float);          // 25.2 MB
    const size_t SB = (size_t)NT * DIM * sizeof(unsigned short); // 12.6 MB

    // Reuse-aware arena (~200 MB total)
    float* xt   = (float*)alloc(SF);             // residual (live to end)
    float* xn   = (float*)alloc(2 * SF);         // xn | e   -> later z (fc1, 384ch)
    float* e    = xn + (size_t)NT * DIM;
    float* z    = xn;
    float* tfb  = (float*)alloc(SF);             // tf -> outS -> xt2
    float* outS = tfb; float* xt2 = tfb;
    unsigned short* xnb = (unsigned short*)alloc(2 * SB);  // xn_bf | e_bf -> xfin f32
    unsigned short* eb  = xnb + (size_t)NT * DIM;
    float* xfin = (float*)xnb;
    unsigned short* enb  = (unsigned short*)alloc(SB);     // emb-ln bf -> mlp-ln bf
    unsigned short* ln2b = enb;
    unsigned short* sxb  = (unsigned short*)alloc(SB);     // sorted xn -> scattered out
    unsigned short* outb = sxb;
    unsigned short* qbb  = (unsigned short*)alloc(SB);     // q -> proj out (t1)
    unsigned short* t1b  = qbb;
    unsigned short* kbb  = (unsigned short*)alloc(2 * SB); // k|v -> (z+zc) bf [NT][384]
    unsigned short* vbb  = kbb + (size_t)NT * DIM;
    unsigned short* sbb  = kbb;
    unsigned long long* kp = (unsigned long long*)alloc((size_t)NT * 8);
    float* rlog   = (float*)alloc((size_t)BATCH * MP * NN * sizeof(float));
    float* pacc   = (float*)alloc((size_t)BATCH * MP * 193 * sizeof(float));
    float* proto0 = (float*)alloc((size_t)BATCH * MP * DIM * sizeof(float));
    float* srk    = (float*)alloc((size_t)BATCH * MP * DIM * sizeof(float));
    float* rsum   = (float*)alloc((size_t)BATCH * MP * DIM * sizeof(float));
    float* pff    = (float*)alloc((size_t)BATCH * MP * QD * sizeof(float));
    unsigned short* Pkb = (unsigned short*)alloc((size_t)BATCH * MP * DIM * 2);
    unsigned short* Pvb = (unsigned short*)alloc((size_t)BATCH * MP * DIM * 2);
    unsigned short* w_emb  = (unsigned short*)alloc(DIM * DIM * 2);
    unsigned short* w_tp   = (unsigned short*)alloc(DIM * QD * 2);
    unsigned short* w_q    = (unsigned short*)alloc(DIM * QD * 2);
    unsigned short* w_k    = (unsigned short*)alloc(DIM * QD * 2);
    unsigned short* w_v    = (unsigned short*)alloc(DIM * DIM * 2);
    unsigned short* w_pr   = (unsigned short*)alloc(DIM * DIM * 2);
    unsigned short* w_c1t  = (unsigned short*)alloc(DIM * DIM * 2);
    unsigned short* w_fc1  = (unsigned short*)alloc(DIM * MDIM * 2);
    unsigned short* w_fc2  = (unsigned short*)alloc(MDIM * DIM * 2);

    const float scale  = 0.0721687836487f;   // 192^-0.5
    const float sc_att = 0.1767766953f;      // 32^-0.5

    auto GEMM = [&](const unsigned short* A, const unsigned short* Bm, const float* bias,
                    const float* resid, float* Cf, unsigned short* Cb, int N_, int K_, int act){
        dim3 grid(N_ / 64, NT / 32);
        k_gemm<<<grid, 256, 0, stream>>>(A, Bm, bias, resid, Cf, Cb, NT, N_, K_, act);
    };

    // weights -> bf16 (incl. conv1^T)
    k_f2b<<<64, 256, 0, stream>>>(emb_w,   w_emb, DIM * DIM);
    k_f2b<<<64, 256, 0, stream>>>(tproj_w, w_tp,  DIM * QD);
    k_f2b<<<64, 256, 0, stream>>>(q_w,     w_q,   DIM * QD);
    k_f2b<<<64, 256, 0, stream>>>(k_w,     w_k,   DIM * QD);
    k_f2b<<<64, 256, 0, stream>>>(v_w,     w_v,   DIM * DIM);
    k_f2b<<<64, 256, 0, stream>>>(proj_w,  w_pr,  DIM * DIM);
    k_f2b<<<64, 256, 0, stream>>>(fc1_w,   w_fc1, DIM * MDIM);
    k_f2b<<<64, 256, 0, stream>>>(fc2_w,   w_fc2, MDIM * DIM);
    k_wtrans<<<(DIM * DIM + 255) / 256, 256, 0, stream>>>(conv1_w, w_c1t);

    // 1) NCHW -> [b][n][c], LayerNorms
    k_transpose<<<dim3(NN / 32, DIM / 32, BATCH), dim3(32, 8), 0, stream>>>(x, xt, DIM, NN);
    k_ln<<<NT / 8, 256, 0, stream>>>(xt, norm_g, norm_b, xn, xnb, NT);
    k_ln<<<NT / 8, 256, 0, stream>>>(xn, embln_g, embln_b, nullptr, enb, NT);

    // 2) e = gelu(ln_emb(xn) @ emb_w + b) ; tf = e @ tproj_w
    GEMM(enb, w_emb, emb_b, nullptr, e, eb, DIM, DIM, 1);
    GEMM(eb, w_tp, nullptr, nullptr, tfb, nullptr, QD, DIM, 0);

    // 3) prototype routing
    k_zero<<<13, 256, 0, stream>>>(pacc, BATCH * MP * 193);
    k_assign<<<dim3(NN / 256, BATCH), 256, 0, stream>>>(e, xn, assign_w, pacc);
    k_proto1<<<BATCH, 256, 0, stream>>>(pacc, pln_g, pln_b, pq, rq_w, rk_w, proto0, srk);
    k_rattn<<<BATCH * MP, 192, 0, stream>>>(srk, e, xn, rlog, rsum, scale);
    k_proto2<<<BATCH, 256, 0, stream>>>(proto0, rsum, rv_w, pln_g, pln_b, gate,
                                        pproj_w, pk_w, pv_w, pff, Pkb, Pvb);
    k_route<<<dim3(NN / 256, BATCH), 256, 0, stream>>>(tfb, pff, kp, scale);

    // 4) stable ascending bitonic sort of packed (key, idx)
    for (int kk = 2; kk <= NN; kk <<= 1)
        for (int j = kk >> 1; j > 0; j >>= 1)
            k_bitonic<<<NT / 256, 256, 0, stream>>>(kp, j, kk);
    k_gather<<<NT * 24 / 256, 256, 0, stream>>>(kp, xnb, sxb);

    // 5) q,k,v ; windowed + proto attention ; scatter back
    GEMM(sxb, w_q, nullptr, nullptr, nullptr, qbb, QD, DIM, 0);
    GEMM(sxb, w_k, nullptr, nullptr, nullptr, kbb, QD, DIM, 0);
    GEMM(sxb, w_v, nullptr, nullptr, nullptr, vbb, DIM, DIM, 0);
    k_attn<<<dim3(NN / GS, NHEAD, BATCH), 128, 0, stream>>>(qbb, kbb, vbb, Pkb, Pvb, outS, sc_att);
    k_scatter<<<NT * DIM / 256, 256, 0, stream>>>(kp, outS, outb);

    // 6) proj, 1x1 conv, residual
    GEMM(outb, w_pr, nullptr, nullptr, nullptr, t1b, DIM, DIM, 0);
    GEMM(t1b, w_c1t, nullptr, xt, xt2, nullptr, DIM, DIM, 0);

    // 7) ConvFFN
    k_ln<<<NT / 8, 256, 0, stream>>>(xt2, mlpln_g, mlpln_b, nullptr, ln2b, NT);
    GEMM(ln2b, w_fc1, fc1_b, nullptr, z, nullptr, MDIM, DIM, 1);
    k_dwconv<<<(size_t)NT * MDIM / 256, 256, 0, stream>>>(z, dw_w, dw_b, sbb);
    GEMM(sbb, w_fc2, fc2_b, xt2, xfin, nullptr, DIM, MDIM, 0);

    // 8) [b][n][c] -> NCHW output
    k_transpose<<<dim3(DIM / 32, NN / 32, BATCH), dim3(32, 8), 0, stream>>>(xfin, out, NN, DIM);
}
